// DeformableConv2d_38268158607369
// MI455X (gfx1250) — matrix operand order repaired and checked by reading, not compiled, not run on hardware
//
#include <hip/hip_runtime.h>
#include <hip/hip_bf16.h>
#include <math.h>

// ---- problem constants ----
#define Bn   4
#define Cn   64
#define On   64
#define Hn   128
#define Wn   128
#define HW   (Hn * Wn)          // 16384
#define KK   9                  // 3x3
#define KDIM (KK * Cn)          // 576 = 18 * 32
#define KSTEPS 18               // K tiles of 32
#define OCH  18                 // offset channels = 2*KK

// LDS A-tile row stride in bytes (576 f16 + pad). 74*16 -> 16B aligned.
#define ROWB 1184

typedef __attribute__((ext_vector_type(16))) _Float16 v16h;
typedef __attribute__((ext_vector_type(8)))  float    v8f;

// ---------------------------------------------------------------------------
// K0a: reorder main weight (O=64,C=64,3,3) f32 -> WMMA B-fragment f16 layout.
// Flat K index: kk = kpos*64 + c.  B frag (32x16, 16-bit):
//   lane 0-15  : column n = lane,    K = kstep*32 + 0..15
//   lane 16-31 : column n = lane-16, K = kstep*32 + 16..31
// Storage: wB[((nt*18 + kstep)*32 + lane)*16 + v], nt = 0..3
// ---------------------------------------------------------------------------
__global__ void wreorder_kernel(const float* __restrict__ wgt,
                                _Float16* __restrict__ wB) {
    int id = blockIdx.x * 256 + threadIdx.x;          // 4*18*32*16 = 36864
    if (id >= 4 * KSTEPS * 32 * 16) return;
    int v    = id & 15;
    int lane = (id >> 4) & 31;
    int ks   = (id >> 9) % KSTEPS;
    int nt   = id / (KSTEPS * 512);
    int K    = ks * 32 + ((lane & 16) ? 16 : 0) + v;  // 0..575
    int o    = nt * 16 + (lane & 15);
    int kpos = K >> 6;
    int c    = K & 63;
    wB[id] = (_Float16)wgt[(o * Cn + c) * KK + kpos];
}

// ---------------------------------------------------------------------------
// K0b: reorder offset weight (18,64,3,3) -> B-frag layout, N padded 18 -> 32.
// Storage: wOB[((nt*18 + kstep)*32 + lane)*16 + v], nt = 0..1
// ---------------------------------------------------------------------------
__global__ void wreorder_off_kernel(const float* __restrict__ ow,
                                    _Float16* __restrict__ wOB) {
    int id = blockIdx.x * 256 + threadIdx.x;          // 2*18*32*16 = 18432
    if (id >= 2 * KSTEPS * 32 * 16) return;
    int v    = id & 15;
    int lane = (id >> 4) & 31;
    int ks   = (id >> 9) % KSTEPS;
    int nt   = id / (KSTEPS * 512);
    int K    = ks * 32 + ((lane & 16) ? 16 : 0) + v;
    int o    = nt * 16 + (lane & 15);                 // 0..31, pad >= 18
    int kpos = K >> 6;
    int c    = K & 63;
    float val = (o < OCH) ? ow[(o * Cn + c) * KK + kpos] : 0.0f;
    wOB[id] = (_Float16)val;
}

// ---------------------------------------------------------------------------
// K1: offset conv as im2col + WMMA.  M-tile = 32 pixels, N = 32 (18 real),
// K = 576.  Grid: 4 b * 128 h * 4 wtiles = 2048 blocks, 128 threads.
// Waves: (mt, nt) = (w>>1, w&1).  offs layout: [b][18][h][w] f32.
// ---------------------------------------------------------------------------
__global__ void __launch_bounds__(128)
offs_conv_wmma_kernel(const float* __restrict__ x,
                      const _Float16* __restrict__ wOB,
                      const float* __restrict__ ob,
                      float* __restrict__ offs) {
    __shared__ uint4 ldsA[32 * ROWB / 16];            // 37888 bytes
    char* ldsB = (char*)ldsA;

    const int tid  = threadIdx.x;
    const int tile = blockIdx.x;
    const int w0   = (tile & 3) * 32;
    const int h    = (tile >> 2) & (Hn - 1);
    const int b    = tile >> 9;

    // ---- im2col fill: 32 rows x 576 f16 (zero-padded borders) ----
    {
        const int p   = tid >> 2;                     // pixel row 0..31
        const int cg  = (tid & 3) * 16;               // 16 channels per thread
        const int wpx = w0 + p;
        const float* xb0 = x + ((size_t)b * Cn + cg) * HW;

        #pragma unroll
        for (int k = 0; k < KK; ++k) {
            int yy = h   + (k / 3) - 1;
            int xx = wpx + (k % 3) - 1;
            bool valid = ((unsigned)yy < (unsigned)Hn) & ((unsigned)xx < (unsigned)Wn);
            union { uint4 q[2]; _Float16 hv[16]; } pk;
            if (valid) {
                const float* xc = xb0 + yy * Wn + xx;
                #pragma unroll
                for (int cc = 0; cc < 16; ++cc) {
                    pk.hv[cc] = (_Float16)xc[(size_t)cc * HW];
                }
            } else {
                pk.q[0] = make_uint4(0u, 0u, 0u, 0u);
                pk.q[1] = make_uint4(0u, 0u, 0u, 0u);
            }
            char* dst = ldsB + p * ROWB + (k * 64 + cg) * 2;   // 32B, 16B aligned
            *(uint4*)dst       = pk.q[0];
            *((uint4*)dst + 1) = pk.q[1];
        }
    }
    __syncthreads();

    // ---- WMMA: wave (mt,nt) does 16x16 over K=576 ----
    const int lane = tid & 31;
    const int wv   = tid >> 5;
    const int mt   = wv >> 1;                         // M subtile 0..1
    const int nt   = wv & 1;                          // N subtile 0..1
    const int m    = lane & 15;
    const int hi   = lane >> 4;

    v8f acc = {};
    const char* aRow = ldsB + (mt * 16 + m) * ROWB + hi * 32;
    const uint4* bPtr = (const uint4*)wOB + ((size_t)nt * KSTEPS) * 64 + lane * 2;

    #pragma unroll
    for (int ks = 0; ks < KSTEPS; ++ks) {
        union { v16h v; uint4 q[2]; } A, Bf;
        A.q[0] = *(const uint4*)(aRow + ks * 64);
        A.q[1] = *(const uint4*)(aRow + ks * 64 + 16);
        const uint4* bp = bPtr + ks * 64;
        Bf.q[0] = bp[0];
        Bf.q[1] = bp[1];
        acc = __builtin_amdgcn_wmma_f32_16x16x32_f16(
                  false, A.v, false, Bf.v, (short)0, acc, false, false);
    }

    // ---- epilogue: only 18 real channels stored ----
    const int o = nt * 16 + m;
    if (o < OCH) {
        const float bv = ob[o];
        union { v8f v; float4 q[2]; } C;
        C.v = acc;
        #pragma unroll
        for (int r = 0; r < 8; ++r) C.v[r] += bv;
        float* op = offs + (((size_t)b * OCH + o) * Hn + h) * Wn
                         + w0 + mt * 16 + hi * 8;
        *(float4*)op       = C.q[0];
        *((float4*)op + 1) = C.q[1];
    }
}

// ---------------------------------------------------------------------------
// K2: fused bilinear sampling + WMMA GEMM.
// Grid: 4096 workgroups (b * 128 h * 8 wtiles), 128 threads (4 waves).
// Each WG: 16 pixels x 64 out-channels, K = 576.
// ---------------------------------------------------------------------------
__global__ void __launch_bounds__(128)
dconv_main_kernel(const float* __restrict__ x,
                  const float* __restrict__ offs,
                  const _Float16* __restrict__ wB,
                  const float* __restrict__ bias,
                  float* __restrict__ out) {
    __shared__ uint4 ldsA[16 * ROWB / 16];            // 18944 bytes
    char* ldsB = (char*)ldsA;

    const int tid  = threadIdx.x;
    const int tile = blockIdx.x;
    const int w0   = (tile & 7) * 16;
    const int h    = (tile >> 3) & (Hn - 1);
    const int b    = tile >> 10;

    // ---- fill phase: sampled A tile (16 x 576) in f16 ----
    {
        const int p   = tid >> 3;                     // pixel row 0..15
        const int cg  = (tid & 7) * 8;                // 8 channels per thread
        const int wpx = w0 + p;
        const float* ofb = offs + (size_t)b * OCH * HW;
        const float* xb0 = x + (size_t)b * Cn * HW;

        for (int k = 0; k < KK; ++k) {
            float dy = ofb[(size_t)(2 * k)     * HW + h * Wn + wpx];
            float dx = ofb[(size_t)(2 * k + 1) * HW + h * Wn + wpx];
            float ys = (float)(h   - 1 + (k / 3)) + dy;
            float xs = (float)(wpx - 1 + (k % 3)) + dx;

            float y0f = floorf(ys), x0f = floorf(xs);
            float wy1 = ys - y0f,   wx1 = xs - x0f;
            float wy0 = 1.0f - wy1, wx0 = 1.0f - wx1;
            int y0 = (int)y0f, x0i = (int)x0f;
            int y1 = y0 + 1,   x1i = x0i + 1;

            float my0 = (y0  >= 0 && y0  <= Hn - 1) ? 1.0f : 0.0f;
            float my1 = (y1  >= 0 && y1  <= Hn - 1) ? 1.0f : 0.0f;
            float mx0 = (x0i >= 0 && x0i <= Wn - 1) ? 1.0f : 0.0f;
            float mx1 = (x1i >= 0 && x1i <= Wn - 1) ? 1.0f : 0.0f;
            int y0c = min(max(y0,  0), Hn - 1), y1c = min(max(y1,  0), Hn - 1);
            int x0c = min(max(x0i, 0), Wn - 1), x1c = min(max(x1i, 0), Wn - 1);

            float w00 = wy0 * wx0 * my0 * mx0;
            float w01 = wy0 * wx1 * my0 * mx1;
            float w10 = wy1 * wx0 * my1 * mx0;
            float w11 = wy1 * wx1 * my1 * mx1;
            int i00 = y0c * Wn + x0c, i01 = y0c * Wn + x1c;
            int i10 = y1c * Wn + x0c, i11 = y1c * Wn + x1c;

            union { uint4 q; _Float16 hv[8]; } pk;
            const float* xc = xb0 + (size_t)cg * HW;
            #pragma unroll
            for (int cc = 0; cc < 8; ++cc) {
                float v = w00 * xc[i00] + w01 * xc[i01]
                        + w10 * xc[i10] + w11 * xc[i11];
                pk.hv[cc] = (_Float16)v;
                xc += HW;
            }
            *(uint4*)(ldsB + p * ROWB + (k * 64 + cg) * 2) = pk.q;
        }
    }
    __syncthreads();

    // ---- WMMA phase ----
    const int lane = tid & 31;
    const int nt   = tid >> 5;                        // N tile 0..3
    const int m    = lane & 15;
    const int hi   = lane >> 4;

    v8f acc = {};
    const char* aRow = ldsB + m * ROWB + hi * 32;
    const uint4* bPtr = (const uint4*)wB + ((size_t)nt * KSTEPS) * 64 + lane * 2;

    #pragma unroll
    for (int ks = 0; ks < KSTEPS; ++ks) {
        union { v16h v; uint4 q[2]; } A, Bf;
        A.q[0] = *(const uint4*)(aRow + ks * 64);
        A.q[1] = *(const uint4*)(aRow + ks * 64 + 16);
        const uint4* bp = bPtr + ks * 64;
        Bf.q[0] = bp[0];
        Bf.q[1] = bp[1];
        acc = __builtin_amdgcn_wmma_f32_16x16x32_f16(
                  false, A.v, false, Bf.v, (short)0, acc, false, false);
    }

    // ---- epilogue ----
    const int o = nt * 16 + m;
    const float bv = bias[o];
    union { v8f v; float4 q[2]; } C;
    C.v = acc;
    #pragma unroll
    for (int r = 0; r < 8; ++r) C.v[r] += bv;

    float* op = out + (((size_t)b * On + o) * Hn + h) * Wn + w0 + hi * 8;
    *(float4*)op       = C.q[0];
    *((float4*)op + 1) = C.q[1];
}

// ---------------------------------------------------------------------------
extern "C" void kernel_launch(void* const* d_in, const int* in_sizes, int n_in,
                              void* d_out, int out_size, void* d_ws, size_t ws_size,
                              hipStream_t stream) {
    const float* x        = (const float*)d_in[0];
    const float* weight   = (const float*)d_in[1];
    const float* bias     = (const float*)d_in[2];
    const float* offset_w = (const float*)d_in[3];
    const float* offset_b = (const float*)d_in[4];
    float* out = (float*)d_out;

    // workspace layout:
    //   [0, 80KB)        reordered main weight f16   (73728 B used)
    //   [80KB, 128KB)    reordered offset weight f16 (36864 B used)
    //   [128KB, ~4.85MB) offset map f32 [4][18][128][128]
    _Float16* wB  = (_Float16*)d_ws;
    _Float16* wOB = (_Float16*)((char*)d_ws + 80 * 1024);
    float*    ofs = (float*)((char*)d_ws + 128 * 1024);

    wreorder_kernel<<<(4 * KSTEPS * 32 * 16) / 256, 256, 0, stream>>>(weight, wB);
    wreorder_off_kernel<<<(2 * KSTEPS * 32 * 16) / 256, 256, 0, stream>>>(offset_w, wOB);
    offs_conv_wmma_kernel<<<Bn * Hn * (Wn / 32), 128, 0, stream>>>(x, wOB, offset_b, ofs);
    dconv_main_kernel<<<Bn * Hn * (Wn / 16), 128, 0, stream>>>(x, ofs, wB, bias, out);
}
